// ALSHConvNet_7198365188563
// MI455X (gfx1250) — matrix operand order repaired and checked by reading, not compiled, not run on hardware
//
#include <hip/hip_runtime.h>
#include <hip/hip_bf16.h>

typedef __attribute__((ext_vector_type(16))) _Float16 v16h;
typedef __attribute__((ext_vector_type(8)))  _Float16 v8h;
typedef __attribute__((ext_vector_type(4)))  _Float16 v4h;
typedef __attribute__((ext_vector_type(8)))  float    v8f;
typedef __attribute__((ext_vector_type(4)))  int      v4i;

#define WMMA_F16(A, B, C) \
  __builtin_amdgcn_wmma_f32_16x16x32_f16(false, (A), false, (B), (short)0, (C), false, false)

// ---- async global->LDS copy (ASYNCcnt path), with safe fallback ------------
#if defined(__has_builtin)
#if __has_builtin(__builtin_amdgcn_global_load_async_to_lds_b128)
#define HAVE_ASYNC_LDS 1
#endif
#endif

#ifdef HAVE_ASYNC_LDS
#define ASYNC_CP16(dst, src)                                               \
  __builtin_amdgcn_global_load_async_to_lds_b128(                          \
      (__attribute__((address_space(1))) v4i*)(src),                       \
      (__attribute__((address_space(3))) v4i*)(dst), 0, 0)
#if __has_builtin(__builtin_amdgcn_s_wait_asynccnt)
#define WAIT_ASYNC() __builtin_amdgcn_s_wait_asynccnt(0)
#else
#define WAIT_ASYNC() asm volatile("s_wait_asynccnt 0x0" ::: "memory")
#endif
#else
#define ASYNC_CP16(dst, src) (*(v8h*)(dst) = *(const v8h*)(src))
#define WAIT_ASYNC()
#endif

// ---------------------------------------------------------------------------
// Per-channel partial sums (deterministic two-stage mean, no atomics).
// grid.x = C*32, block = 256, chunk = 64 batches (B = 2048 always).
// ---------------------------------------------------------------------------
template <typename T>
__global__ __launch_bounds__(256) void chan_mean_partial(
    const T* __restrict__ x, float* __restrict__ part, int C, int HW) {
  __shared__ float red[256];
  const int c = blockIdx.x >> 5, chunk = blockIdx.x & 31;
  const int tid = threadIdx.x;
  float s = 0.f;
  const int total = 64 * HW;
  for (int i = tid; i < total; i += 256) {
    int b = chunk * 64 + i / HW;
    int hw = i % HW;
    s += (float)x[((size_t)b * C + c) * HW + hw];
  }
  red[tid] = s; __syncthreads();
  for (int st = 128; st > 0; st >>= 1) {
    if (tid < st) red[tid] += red[tid + st];
    __syncthreads();
  }
  if (tid == 0) part[c * 32 + chunk] = red[0];
}

// NHWC variant: element (b,hw,c) at x[(b*HW+hw)*Cs + c]
__global__ __launch_bounds__(256) void chan_mean_partial_nhwc(
    const _Float16* __restrict__ x, float* __restrict__ part, int Cs, int HW) {
  __shared__ float red[256];
  const int c = blockIdx.x >> 5, chunk = blockIdx.x & 31;
  const int tid = threadIdx.x;
  float s = 0.f;
  const int total = 64 * HW;
  for (int i = tid; i < total; i += 256) {
    int b = chunk * 64 + i / HW;
    int hw = i % HW;
    s += (float)x[((size_t)b * HW + hw) * Cs + c];
  }
  red[tid] = s; __syncthreads();
  for (int st = 128; st > 0; st >>= 1) {
    if (tid < st) red[tid] += red[tid + st];
    __syncthreads();
  }
  if (tid == 0) part[c * 32 + chunk] = red[0];
}

// ---------------------------------------------------------------------------
// ALSH mask + masked f16 weight-fragment packing (one block).
// B fragment layout: Bpack[((ntile*ksteps+ks)*32+lane)*16 + j],
//   half = lane>>4, n = (lane&15)+16*ntile, K = ks*32 + 16*half + j.
// packMode 0: K linear over (ci*25 + kh*5 + kw)
// packMode 1: K = tap*16 + ci  (tap = kh*5+kw, Cpad=16)     (conv2)
// packMode 2: K = tap*32 + ci  (Cpad=32)                    (conv3)
// packMode 3: K = tap*8  + ci  (Cpad=8)                     (conv1)
// ---------------------------------------------------------------------------
#define MAXC 20
__global__ __launch_bounds__(256) void prep_kernel(
    const float* __restrict__ W, const float* __restrict__ A,
    const float* __restrict__ part, float invCount,
    int outch, int inch, int L, int Kh, int ksteps, int ntiles, int packMode,
    _Float16* __restrict__ Bpack) {
  const int D = inch * 25, Dp = D + 3;
  __shared__ float cm[MAXC], norm2[MAXC], terms[MAXC * 3], maskv[MAXC];
  __shared__ float sScale, qInv;
  __shared__ int codeQ[4];
  __shared__ unsigned char hQ[40];
  __shared__ unsigned char hW[32 * MAXC];
  const int tid = threadIdx.x;

  if (tid < inch) {
    float s = 0.f;
    for (int j = 0; j < 32; ++j) s += part[tid * 32 + j];
    cm[tid] = s * invCount;
  }
  if (tid < outch) {
    float s = 0.f;
    const float* w = W + tid * D;
    for (int d = 0; d < D; ++d) s += w[d] * w[d];
    norm2[tid] = s;
  }
  __syncthreads();
  if (tid == 0) {
    float mx = 0.f;
    for (int o = 0; o < outch; ++o) mx = fmaxf(mx, norm2[o]);
    sScale = 0.9999f / sqrtf(mx);
    float qs = 0.f;
    for (int ci = 0; ci < inch; ++ci) qs += cm[ci] * cm[ci];
    qInv = 1.0f / (sqrtf(25.0f * qs) + 1e-8f);
  }
  __syncthreads();
  if (tid < outch) {
    float n2 = sScale * sScale * norm2[tid];
    float n4 = n2 * n2;
    terms[tid * 3 + 0] = 0.5f - n2;
    terms[tid * 3 + 1] = 0.5f - n4;
    terms[tid * 3 + 2] = 0.5f - n4 * n4;
  }
  if (tid < L * Kh) {                      // query hash bits
    const float* a = A + (size_t)tid * Dp;
    float dot = 0.f;
    for (int d = 0; d < D; ++d) dot += a[d] * cm[d / 25];
    hQ[tid] = (dot * qInv > 0.f) ? 1 : 0;
  }
  __syncthreads();
  if (tid < L) {
    int code = 0;
    for (int kk = 0; kk < Kh; ++kk) code += (int)hQ[tid * Kh + kk] << kk;
    codeQ[tid] = code % 25;
  }
  for (int idx = tid; idx < L * Kh * outch; idx += 256) {   // weight hash bits
    int o = idx % outch, lk = idx / outch;
    const float* a = A + (size_t)lk * Dp;
    const float* w = W + (size_t)o * D;
    float dot = 0.f;
    for (int d = 0; d < D; ++d) dot += a[d] * w[d];
    dot *= sScale;
    dot += a[D] * terms[o*3] + a[D+1] * terms[o*3+1] + a[D+2] * terms[o*3+2];
    hW[lk * outch + o] = (dot > 0.f) ? 1 : 0;
  }
  __syncthreads();
  if (tid < outch) {
    float m = 0.f;
    for (int l = 0; l < L; ++l) {
      int code = 0;
      for (int kk = 0; kk < Kh; ++kk) code += (int)hW[(l*Kh+kk)*outch + tid] << kk;
      if (code % 25 == codeQ[l]) m = 1.f;
    }
    maskv[tid] = m;
  }
  __syncthreads();
  const int total = ntiles * ksteps * 512;
  for (int idx = tid; idx < total; idx += 256) {
    int j = idx & 15, lane = (idx >> 4) & 31, t = idx >> 9;
    int ks = t % ksteps, ntile = t / ksteps;
    int half = lane >> 4, n = (lane & 15) + 16 * ntile;
    int K = ks * 32 + 16 * half + j;
    float v = 0.f;
    if (n < outch) {
      if (packMode == 0) {
        if (K < D) v = W[(size_t)n * D + K] * maskv[n];
      } else if (packMode == 1) {
        int T = K >> 4, ci = K & 15;
        if (T < 25 && ci < inch) v = W[(size_t)n * D + ci * 25 + T] * maskv[n];
      } else if (packMode == 2) {
        int T = K >> 5, ci = K & 31;
        if (T < 25 && ci < inch) v = W[(size_t)n * D + ci * 25 + T] * maskv[n];
      } else {
        int T = K >> 3, ci = K & 7;
        if (T < 25 && ci < inch) v = W[(size_t)n * D + ci * 25 + T] * maskv[n];
      }
    }
    Bpack[idx] = (_Float16)v;
  }
}

// tap offset (halfwords) in a [12][36][8] patch, dummy taps clamped to 24
__device__ __forceinline__ constexpr int tap_off36_8(int t) {
  int tc = (t < 25) ? t : 24;
  return ((tc / 5) * 36 + (tc % 5)) * 8;
}

// ---------------------------------------------------------------------------
// Conv1: x(2048,3,32,32)f32 -> h1 NHWC (2048,16,16,16ch)f16.
// Tap-major K with Cpad=8: 25 taps x 8 -> 7 ksteps, A-frag = 2x ds_load_b128.
// Block = (b,p4), 8 waves; wave = (oh pair, ow half). Fused mask/ReLU/pool.
// ---------------------------------------------------------------------------
__global__ __launch_bounds__(256) void conv1_kernel(
    const float* __restrict__ x, const _Float16* __restrict__ Bpack,
    _Float16* __restrict__ h1) {
  __shared__ __align__(16) _Float16 patch[12 * 36 * 8];   // [row][col][ci(8)]
  const int b = blockIdx.x >> 3, p4 = blockIdx.x & 7;
  const int tid = threadIdx.x;
  for (int i = tid; i < 12 * 36 * 8; i += 256) {
    int ci = i & 7, pc = i >> 3;
    int r = pc / 36, cp = pc % 36;
    int ih = p4 * 8 - 2 + r, iw = cp - 2;
    float v = 0.f;
    if (ci < 3 && ih >= 0 && ih < 32 && iw >= 0 && iw < 32)
      v = x[(((size_t)b * 3 + ci) << 10) + (ih << 5) + iw];
    patch[i] = (_Float16)v;
  }
  __syncthreads();
  const int lane = tid & 31, wid = tid >> 5;
  const int pp = wid >> 1, w16 = wid & 1;
  const int half = lane >> 4, m = lane & 15;
  const int ow = w16 * 16 + m;
  const int base0 = ((2 * pp) * 36 + ow) * 8;
  const int base1 = base0 + 36 * 8;
  v8f acc0 = {}, acc1 = {};
  #pragma unroll
  for (int ks = 0; ks < 7; ++ks) {
    const int ol = half ? tap_off36_8(4 * ks + 2) : tap_off36_8(4 * ks);
    const int oh = half ? tap_off36_8(4 * ks + 3) : tap_off36_8(4 * ks + 1);
    v16h bf = *(const v16h*)(Bpack + (size_t)(ks * 32 + lane) * 16);
    v16h a0, a1;
    *((v8h*)&a0)     = *(const v8h*)&patch[base0 + ol];
    *((v8h*)&a0 + 1) = *(const v8h*)&patch[base0 + oh];
    *((v8h*)&a1)     = *(const v8h*)&patch[base1 + ol];
    *((v8h*)&a1 + 1) = *(const v8h*)&patch[base1 + oh];
    acc0 = WMMA_F16(a0, bf, acc0);
    acc1 = WMMA_F16(a1, bf, acc1);
  }
  const int ph = p4 * 4 + pp, pwb = w16 * 8 + half * 4;
  #pragma unroll
  for (int i = 0; i < 4; ++i) {
    float p = fmaxf(fmaxf(acc0[2*i], acc0[2*i+1]), fmaxf(acc1[2*i], acc1[2*i+1]));
    h1[((((size_t)b * 16 + ph) * 16) + pwb + i) * 16 + m] = (_Float16)fmaxf(p, 0.f);
  }
}

// ---------------------------------------------------------------------------
// Conv2: h1 NHWC(2048,16,16,16) -> h2 NHWC32 (2048,8,8,32). Tap-major K:
// 25 taps x 16ci = 400 -> 13 ksteps (2 taps each). A-frag = 2x ds_load_b128.
// Block = (b, ntile), 8 waves = 8 oh pairs.
// ---------------------------------------------------------------------------
__global__ __launch_bounds__(256) void conv2_kernel(
    const _Float16* __restrict__ h1, const _Float16* __restrict__ Bpack,
    _Float16* __restrict__ h2) {
  __shared__ __align__(16) _Float16 patch[20 * 20 * 16];   // [row][col][ci]
  const int b = blockIdx.x >> 1, ntile = blockIdx.x & 1;
  const int tid = threadIdx.x;
  for (int p = tid; p < 400; p += 256) {
    int rr = p / 20, cc = p % 20;
    int ih = rr - 2, iw = cc - 2;
    _Float16* dst = &patch[p * 16];
    if (ih >= 0 && ih < 16 && iw >= 0 && iw < 16) {
      const _Float16* src = h1 + (((size_t)b * 16 + ih) * 16 + iw) * 16;
      ASYNC_CP16(dst, src);
      ASYNC_CP16(dst + 8, src + 8);
    } else {
      v8h z = {};
      *(v8h*)dst = z; *(v8h*)(dst + 8) = z;
    }
  }
  WAIT_ASYNC();
  __syncthreads();
  const int lane = tid & 31, pp = tid >> 5;
  const int half = lane >> 4, m = lane & 15, off = 8 * half;
  const int base0 = ((2 * pp) * 20 + m) * 16 + off;
  const int base1 = base0 + 20 * 16;
  v8f acc0 = {}, acc1 = {};
  const _Float16* bb = Bpack + (size_t)ntile * (13 * 512);
  #pragma unroll
  for (int ks = 0; ks < 13; ++ks) {
    const int t0 = 2 * ks;
    const int t1 = (2 * ks + 1 < 25) ? (2 * ks + 1) : 24;  // pad tap: B==0
    const int o0 = (half ? ((t1 / 5) * 20 + (t1 % 5)) * 16 : ((t0 / 5) * 20 + (t0 % 5)) * 16) - off;
    const int o1 = o0 + 8;
    v16h bf = *(const v16h*)(bb + (size_t)(ks * 32 + lane) * 16);
    v16h a0, a1;
    *((v8h*)&a0)     = *(const v8h*)&patch[base0 + o0];
    *((v8h*)&a0 + 1) = *(const v8h*)&patch[base0 + o1];
    *((v8h*)&a1)     = *(const v8h*)&patch[base1 + o0];
    *((v8h*)&a1 + 1) = *(const v8h*)&patch[base1 + o1];
    acc0 = WMMA_F16(a0, bf, acc0);
    acc1 = WMMA_F16(a1, bf, acc1);
  }
  const int neff = m + 16 * ntile;   // 0..31; ch >= 20 computed as exact zeros
  #pragma unroll
  for (int i = 0; i < 4; ++i) {
    float p = fmaxf(fmaxf(acc0[2*i], acc0[2*i+1]), fmaxf(acc1[2*i], acc1[2*i+1]));
    h2[((((size_t)b * 8 + pp) * 8) + half * 4 + i) * 32 + neff] = (_Float16)fmaxf(p, 0.f);
  }
}

// ---------------------------------------------------------------------------
// Conv3: h2 NHWC32 -> h3 NCHW (2048,20,4,4). Tap-major K: 25 taps x 32ci ->
// 25 ksteps, A-frag = 2x ds_load_b128. oh-pool via ds_swizzle SWAPX16.
// Block = b, 4 waves (128 thr) = (q, ntile).
// ---------------------------------------------------------------------------
__global__ __launch_bounds__(128) void conv3_kernel(
    const _Float16* __restrict__ h2, const _Float16* __restrict__ Bpack,
    _Float16* __restrict__ h3) {
  __shared__ __align__(16) _Float16 patch[12 * 12 * 32];   // [row][col][ci]
  const int b = blockIdx.x;
  const int tid = threadIdx.x;
  for (int p = tid; p < 144; p += 128) {
    int rr = p / 12, cc = p % 12;
    int ih = rr - 2, iw = cc - 2;
    _Float16* dst = &patch[p * 32];
    if (ih >= 0 && ih < 8 && iw >= 0 && iw < 8) {
      const _Float16* src = h2 + (((size_t)b * 8 + ih) * 8 + iw) * 32;
      ASYNC_CP16(dst, src);
      ASYNC_CP16(dst + 8,  src + 8);
      ASYNC_CP16(dst + 16, src + 16);
      ASYNC_CP16(dst + 24, src + 24);
    } else {
      v8h z = {};
      *(v8h*)dst = z; *(v8h*)(dst + 8) = z;
      *(v8h*)(dst + 16) = z; *(v8h*)(dst + 24) = z;
    }
  }
  WAIT_ASYNC();
  __syncthreads();
  const int lane = tid & 31, wid = tid >> 5;
  const int q = wid >> 1, ntile = wid & 1;
  const int half = lane >> 4, m = lane & 15, off = 16 * half;
  const int base0 = ((4 * q + (m >> 3)) * 12 + (m & 7)) * 32 + off;
  const int base1 = base0 + 2 * 12 * 32;     // oh + 2
  v8f acc0 = {}, acc1 = {};
  const _Float16* bb = Bpack + (size_t)ntile * (25 * 512);
  #pragma unroll
  for (int T = 0; T < 25; ++T) {
    const int to = ((T / 5) * 12 + (T % 5)) * 32;
    v16h bf = *(const v16h*)(bb + (size_t)(T * 32 + lane) * 16);
    v16h a0, a1;
    *((v8h*)&a0)     = *(const v8h*)&patch[base0 + to];
    *((v8h*)&a0 + 1) = *(const v8h*)&patch[base0 + to + 8];
    *((v8h*)&a1)     = *(const v8h*)&patch[base1 + to];
    *((v8h*)&a1 + 1) = *(const v8h*)&patch[base1 + to + 8];
    acc0 = WMMA_F16(a0, bf, acc0);
    acc1 = WMMA_F16(a1, bf, acc1);
  }
  const int neff = m + 16 * ntile;
  v4h o0, o1;
  #pragma unroll
  for (int i = 0; i < 4; ++i) {
    float q0 = fmaxf(acc0[2*i], acc0[2*i+1]);
    float q1 = fmaxf(acc1[2*i], acc1[2*i+1]);
    float s0 = __builtin_bit_cast(float,
        __builtin_amdgcn_ds_swizzle(__builtin_bit_cast(int, q0), 0x401f)); // SWAPX16
    float s1 = __builtin_bit_cast(float,
        __builtin_amdgcn_ds_swizzle(__builtin_bit_cast(int, q1), 0x401f));
    o0[i] = (_Float16)fmaxf(fmaxf(q0, s0), 0.f);
    o1[i] = (_Float16)fmaxf(fmaxf(q1, s1), 0.f);
  }
  if (half == 0 && neff < 20) {
    _Float16* pr = h3 + (((size_t)b * 20 + neff) * 4 + 2 * q) * 4;
    *(v4h*)pr = o0;
    *(v4h*)(pr + 4) = o1;
  }
}

// ---------------------------------------------------------------------------
// FC head: pack Wout (10x320) f16 B-fragments, then h3 @ Wout^T + bout.
// ---------------------------------------------------------------------------
__global__ __launch_bounds__(256) void packfc_kernel(
    const float* __restrict__ Wout, _Float16* __restrict__ Bp) {
  for (int idx = threadIdx.x; idx < 10 * 512; idx += 256) {
    int j = idx & 15, lane = (idx >> 4) & 31, ks = idx >> 9;
    int half = lane >> 4, n = lane & 15;
    int K = ks * 32 + 16 * half + j;
    float v = (n < 10) ? Wout[(size_t)n * 320 + K] : 0.f;
    Bp[idx] = (_Float16)v;
  }
}

__global__ __launch_bounds__(256) void fc_kernel(
    const _Float16* __restrict__ h3, const _Float16* __restrict__ Bpack,
    const float* __restrict__ bout, float* __restrict__ out) {
  const int tid = threadIdx.x, lane = tid & 31, wid = tid >> 5;
  const int half = lane >> 4, m = lane & 15;
  const int mbase = (blockIdx.x * 8 + wid) * 16;
  const _Float16* row = h3 + (size_t)(mbase + m) * 320;
  v8f acc = {};
  #pragma unroll
  for (int ks = 0; ks < 10; ++ks) {
    const int kb = ks * 32;
    v16h a;
    *((v8h*)&a)     = *(const v8h*)(row + kb + 16 * half);
    *((v8h*)&a + 1) = *(const v8h*)(row + kb + 16 * half + 8);
    v16h bf = *(const v16h*)(Bpack + (size_t)(ks * 32 + lane) * 16);
    acc = WMMA_F16(a, bf, acc);
  }
  if (m < 10) {
    const float bb = bout[m];
    #pragma unroll
    for (int r = 0; r < 8; ++r) {
      int bi = mbase + r + 8 * half;
      out[(size_t)bi * 10 + m] = acc[r] + bb;
    }
  }
}

// ---------------------------------------------------------------------------
// Workspace layout (bytes)
// ---------------------------------------------------------------------------
static constexpr size_t OFF_PART = 0;          //  20*32 f32
static constexpr size_t OFF_BP1  = 4096;       //  7*512 f16 = 7168 B
static constexpr size_t OFF_BP2  = 12288;      //  2*13*512 f16 = 26624 B
static constexpr size_t OFF_BP3  = 40960;      //  2*25*512 f16 = 51200 B
static constexpr size_t OFF_BPFC = 94208;      //  10*512 f16 = 10240 B
static constexpr size_t OFF_H1   = 106496;     //  2048*16*16*16 f16 = 16 MiB (NHWC)
static constexpr size_t OFF_H2   = 16883712;   //  2048*8*8*32  f16 = 8 MiB (NHWC32)
static constexpr size_t OFF_H3   = 25272320;   //  2048*20*4*4  f16 (NCHW)

extern "C" void kernel_launch(void* const* d_in, const int* in_sizes, int n_in,
                              void* d_out, int out_size, void* d_ws, size_t ws_size,
                              hipStream_t stream) {
  const float* x    = (const float*)d_in[0];
  const float* W1   = (const float*)d_in[1];
  const float* A1   = (const float*)d_in[2];
  const float* W2   = (const float*)d_in[3];
  const float* A2   = (const float*)d_in[4];
  const float* W3   = (const float*)d_in[5];
  const float* A3   = (const float*)d_in[6];
  const float* Wout = (const float*)d_in[7];
  const float* bout = (const float*)d_in[8];
  float* out = (float*)d_out;

  char* ws = (char*)d_ws;
  float*    part = (float*)(ws + OFF_PART);
  _Float16* BP1  = (_Float16*)(ws + OFF_BP1);
  _Float16* BP2  = (_Float16*)(ws + OFF_BP2);
  _Float16* BP3  = (_Float16*)(ws + OFF_BP3);
  _Float16* BPFC = (_Float16*)(ws + OFF_BPFC);
  _Float16* H1   = (_Float16*)(ws + OFF_H1);
  _Float16* H2   = (_Float16*)(ws + OFF_H2);
  _Float16* H3   = (_Float16*)(ws + OFF_H3);

  // Layer 1
  chan_mean_partial<float><<<dim3(3 * 32), dim3(256), 0, stream>>>(x, part, 3, 1024);
  prep_kernel<<<dim3(1), dim3(256), 0, stream>>>(W1, A1, part, 1.f / (2048.f * 1024.f),
                                                 16, 3, 2, 6, 7, 1, 3, BP1);
  conv1_kernel<<<dim3(2048 * 8), dim3(256), 0, stream>>>(x, BP1, H1);
  // Layer 2
  chan_mean_partial_nhwc<<<dim3(16 * 32), dim3(256), 0, stream>>>(H1, part, 16, 256);
  prep_kernel<<<dim3(1), dim3(256), 0, stream>>>(W2, A2, part, 1.f / (2048.f * 256.f),
                                                 20, 16, 2, 6, 13, 2, 1, BP2);
  conv2_kernel<<<dim3(2048 * 2), dim3(256), 0, stream>>>(H1, BP2, H2);
  // Layer 3
  chan_mean_partial_nhwc<<<dim3(20 * 32), dim3(256), 0, stream>>>(H2, part, 32, 64);
  prep_kernel<<<dim3(1), dim3(256), 0, stream>>>(W3, A3, part, 1.f / (2048.f * 64.f),
                                                 20, 20, 3, 10, 25, 2, 2, BP3);
  conv3_kernel<<<dim3(2048), dim3(128), 0, stream>>>(H2, BP3, H3);
  // FC head
  packfc_kernel<<<dim3(1), dim3(256), 0, stream>>>(Wout, BPFC);
  fc_kernel<<<dim3(16), dim3(256), 0, stream>>>(H3, BPFC, bout, out);
}
